// GAT_6399501271417
// MI455X (gfx1250) — matrix operand order repaired and checked by reading, not compiled, not run on hardware
//
#include <hip/hip_runtime.h>
#include <hip/hip_bf16.h>
#include <math.h>

typedef __attribute__((ext_vector_type(16))) __bf16 v16bf;
typedef __attribute__((ext_vector_type(8)))  __bf16 v8bf;
typedef __attribute__((ext_vector_type(8)))  float  v8f;

#define NNODES 65536
#define NEDGES 524288
#define NGRP   256
#define BN_EPS 1e-5f

// ---------- float atomic-max encoding (monotone map to unsigned) ----------
__device__ __forceinline__ unsigned fenc(float f) {
  unsigned u = __float_as_uint(f);
  return (u & 0x80000000u) ? ~u : (u | 0x80000000u);
}
__device__ __forceinline__ float fdec(unsigned u) {
  return (u & 0x80000000u) ? __uint_as_float(u & 0x7FFFFFFFu) : __uint_as_float(~u);
}

// ---------- fills ----------
__global__ void fill_f32(float* p, float v, int n) {
  int i = blockIdx.x * blockDim.x + threadIdx.x;
  if (i < n) p[i] = v;
}
__global__ void fill_u32(unsigned* p, unsigned v, int n) {
  int i = blockIdx.x * blockDim.x + threadIdx.x;
  if (i < n) p[i] = v;
}

// ---------- fp32 -> bf16 elementwise ----------
__global__ void cvt_bf16(const float* __restrict__ in, __bf16* __restrict__ out, int n) {
  int i = blockIdx.x * blockDim.x + threadIdx.x;
  if (i < n) out[i] = (__bf16)in[i];
}

// ---------- pack W [K,Cout] fp32 into WMMA B-fragment order (bf16) ----------
// layout: idx = ((tn*(K/32) + kb)*32 + lane)*16 + e, with
// half = lane>>4, col = tn*16 + (lane&15), k = kb*32 + (e>>3)*16 + half*8 + (e&7)
__global__ void pack_W(const float* __restrict__ W, __bf16* __restrict__ out, int K, int Cout)
{
  const int idx = blockIdx.x * blockDim.x + threadIdx.x;
  if (idx >= K * Cout) return;
  const int e    = idx & 15;
  const int lane = (idx >> 4) & 31;
  const int t    = idx >> 9;
  const int KB   = K >> 5;
  const int kb   = t % KB;
  const int tn   = t / KB;
  const int half = lane >> 4;
  const int col  = tn * 16 + (lane & 15);
  const int k    = kb * 32 + (e >> 3) * 16 + half * 8 + (e & 7);
  out[idx] = (__bf16)W[(size_t)k * Cout + col];
}

// ---------- dual WMMA GEMM (templated, unrolled, double-buffered) ----------
// A [M,K] bf16 row-major; PBL/PBR pre-packed bf16 weight fragments; fp32 bias; fp32 out.
// One wave per 16x16 output tile (pair). blockDim = 32. grid = (M/16, COUT/16).
template<int K, int COUT, bool DOR>
__global__ void wmma_gemm_t(const __bf16* __restrict__ A,
                            const __bf16* __restrict__ PBL, const float* __restrict__ BL,
                            const __bf16* __restrict__ PBR, const float* __restrict__ BR,
                            float* __restrict__ OL, float* __restrict__ OR_)
{
  constexpr int KB = K / 32;
  const int lane = threadIdx.x & 31;
  const int tm = blockIdx.x;
  const int tn = blockIdx.y;
  const int half = lane >> 4;
  const int l15  = lane & 15;
  const int col  = tn * 16 + l15;

  // A: lane<16 -> K 0..7 / 16..23 ; lane>=16 -> 8..15 / 24..31 (16B-aligned runs)
  const __bf16* arow = A + (size_t)(tm * 16 + l15) * K + half * 8;
  const __bf16* pbl  = PBL + ((size_t)tn * KB * 32 + lane) * 16;
  const __bf16* pbr  = PBR + ((size_t)tn * KB * 32 + lane) * 16;

  v8f cl = {};
  v8f cr = {};

  // software pipeline: preload fragment 0
  v8bf  a0  = *(const v8bf*)(arow);
  v8bf  a1  = *(const v8bf*)(arow + 16);
  v16bf bfl = *(const v16bf*)(pbl);
  v16bf bfr;
  if (DOR) bfr = *(const v16bf*)(pbr);

#pragma unroll
  for (int kb = 0; kb < KB; ++kb) {
    v8bf  na0, na1;
    v16bf nbl, nbr;
    if (kb + 1 < KB) {                       // issue next-iteration loads first
      na0 = *(const v8bf*)(arow + (kb + 1) * 32);
      na1 = *(const v8bf*)(arow + (kb + 1) * 32 + 16);
      nbl = *(const v16bf*)(pbl + (kb + 1) * 512);
      if (DOR) nbr = *(const v16bf*)(pbr + (kb + 1) * 512);
    }
    v16bf a;
#pragma unroll
    for (int e = 0; e < 8; ++e) { a[e] = a0[e]; a[e + 8] = a1[e]; }
    cl = __builtin_amdgcn_wmma_f32_16x16x32_bf16(false, a, false, bfl, (short)0, cl, false, false);
    if (DOR)
      cr = __builtin_amdgcn_wmma_f32_16x16x32_bf16(false, a, false, bfr, (short)0, cr, false, false);
    if (kb + 1 < KB) {
      a0 = na0; a1 = na1; bfl = nbl;
      if (DOR) bfr = nbr;
    }
  }

  // C/D layout: VGPR r -> row (r + half*8), col = lane&15
  const int rbase = tm * 16 + half * 8;
  const float addL = BL[col];
#pragma unroll
  for (int r = 0; r < 8; ++r)
    OL[(size_t)(rbase + r) * COUT + col] = cl[r] + addL;
  if (DOR) {
    const float addR = BR[col];
#pragma unroll
    for (int r = 0; r < 8; ++r)
      OR_[(size_t)(rbase + r) * COUT + col] = cr[r] + addR;
  }
}

// ---------- attention logits + segment max (wave per edge) ----------
__global__ void edge_logit_max(const float* __restrict__ XL, const float* __restrict__ XR,
                               const float* __restrict__ att,
                               const int* __restrict__ esrc, const int* __restrict__ edst,
                               int E_, int N_, int C,
                               float* __restrict__ Elog, unsigned* __restrict__ MAXV)
{
  const int wid  = blockIdx.x * (blockDim.x >> 5) + (threadIdx.x >> 5);
  const int lane = threadIdx.x & 31;
  const int EA = E_ + N_;
  if (wid >= EA) return;
  const int s = (wid < E_) ? esrc[wid] : (wid - E_);
  const int d = (wid < E_) ? edst[wid] : (wid - E_);
  float sum = 0.f;
  for (int c = lane; c < C; c += 32) {
    float v = XL[(size_t)s * C + c] + XR[(size_t)d * C + c];
    v = (v > 0.f) ? v : 0.2f * v;          // leaky_relu(., 0.2)
    sum += att[c] * v;
  }
#pragma unroll
  for (int off = 16; off > 0; off >>= 1) sum += __shfl_xor(sum, off, 32);
  if (lane == 0) {
    Elog[wid] = sum;
    atomicMax(&MAXV[d], fenc(sum));
  }
}

// ---------- exp(e - max) and denominator ----------
__global__ void edge_exp_den(const int* __restrict__ edst,
                             int E_, int N_,
                             const unsigned* __restrict__ MAXV,
                             float* __restrict__ Elog, float* __restrict__ DEN)
{
  const int i = blockIdx.x * blockDim.x + threadIdx.x;
  const int EA = E_ + N_;
  if (i >= EA) return;
  const int d = (i < E_) ? edst[i] : (i - E_);
  const float ex = __expf(Elog[i] - fdec(MAXV[d]));
  Elog[i] = ex;
  atomicAdd(&DEN[d], ex);
}

// ---------- H[n,c] = bias[c] ----------
__global__ void init_bias(float* __restrict__ H, const float* __restrict__ b, int C, int total)
{
  const int i = blockIdx.x * blockDim.x + threadIdx.x;
  if (i < total) H[i] = b[i % C];
}

// ---------- weighted scatter-add: H[dst] += alpha * XL[src] ----------
__global__ void edge_scatter(const float* __restrict__ XL,
                             const float* __restrict__ EXv, const float* __restrict__ DEN,
                             const int* __restrict__ esrc, const int* __restrict__ edst,
                             int E_, int N_, int C, float* __restrict__ H)
{
  const int idx = blockIdx.x * blockDim.x + threadIdx.x;
  const int total = (E_ + N_) * C;
  if (idx >= total) return;
  const int e = idx / C;
  const int c = idx - e * C;
  const int s = (e < E_) ? esrc[e] : (e - E_);
  const int d = (e < E_) ? edst[e] : (e - E_);
  const float alpha = EXv[e] / DEN[d];
  atomicAdd(&H[(size_t)d * C + c], alpha * XL[(size_t)s * C + c]);
}

// ---------- ReLU in place ----------
__global__ void relu_inplace(float* __restrict__ H, int n)
{
  const int i = blockIdx.x * blockDim.x + threadIdx.x;
  if (i < n) H[i] = fmaxf(H[i], 0.f);
}

// ---------- per-channel mean / biased var (one block per channel) ----------
__global__ void channel_stats(const float* __restrict__ H, int rows, int C,
                              float* __restrict__ mean, float* __restrict__ var)
{
  __shared__ float s1[256];
  __shared__ float s2[256];
  const int c = blockIdx.x;
  float s = 0.f, q = 0.f;
  for (int r = threadIdx.x; r < rows; r += blockDim.x) {
    const float v = H[(size_t)r * C + c];
    s += v; q += v * v;
  }
  s1[threadIdx.x] = s; s2[threadIdx.x] = q;
  __syncthreads();
  for (int o = blockDim.x >> 1; o > 0; o >>= 1) {
    if (threadIdx.x < (unsigned)o) { s1[threadIdx.x] += s1[threadIdx.x + o]; s2[threadIdx.x] += s2[threadIdx.x + o]; }
    __syncthreads();
  }
  if (threadIdx.x == 0) {
    const float mu = s1[0] / (float)rows;
    mean[c] = mu;
    var[c]  = s2[0] / (float)rows - mu * mu;
  }
}

// ---------- BN normalize in place; optionally pool into P[batch, coff(+coff2)+c] ----------
__global__ void bn_norm_pool(float* __restrict__ H, int rows, int C,
                             const float* __restrict__ mean, const float* __restrict__ var,
                             const float* __restrict__ g, const float* __restrict__ b,
                             const int* __restrict__ batch, float* __restrict__ P,
                             int pstride, int coff, int coff2)
{
  const int idx = blockIdx.x * blockDim.x + threadIdx.x;
  if (idx >= rows * C) return;
  const int n = idx / C;
  const int c = idx - n * C;
  const float v = g[c] * (H[idx] - mean[c]) * rsqrtf(var[c] + BN_EPS) + b[c];
  H[idx] = v;
  if (P != nullptr) {
    const int gi = batch[n];
    atomicAdd(&P[(size_t)gi * pstride + coff + c], v);
    if (coff2 >= 0) atomicAdd(&P[(size_t)gi * pstride + coff2 + c], v);
  }
}

// ---------- tiny head: Z = H @ W(128x3) + b ----------
__global__ void lin2_kernel(const float* __restrict__ Hin, const float* __restrict__ W,
                            const float* __restrict__ b, float* __restrict__ Z)
{
  const int idx = blockIdx.x * blockDim.x + threadIdx.x;
  if (idx >= NGRP * 3) return;
  const int gi = idx / 3, j = idx - gi * 3;
  float s = b[j];
  for (int k = 0; k < 128; ++k) s += Hin[gi * 128 + k] * W[k * 3 + j];
  Z[idx] = s;
}

// ---------- sigmoid + log_softmax to d_out ----------
__global__ void final_out(const float* __restrict__ Z, float* __restrict__ out)
{
  const int gi = blockIdx.x * blockDim.x + threadIdx.x;
  if (gi >= NGRP) return;
  const float z0 = Z[gi * 3 + 0], z1 = Z[gi * 3 + 1], z2 = Z[gi * 3 + 2];
  out[gi * 3 + 0] = 1.f / (1.f + __expf(-z0));
  out[gi * 3 + 1] = 1.f / (1.f + __expf(-z1));
  out[gi * 3 + 2] = 1.f / (1.f + __expf(-z2));
  const float m = fmaxf(z0, fmaxf(z1, z2));
  const float l = logf(__expf(z0 - m) + __expf(z1 - m) + __expf(z2 - m));
  out[NGRP * 3 + gi * 3 + 0] = z0 - m - l;
  out[NGRP * 3 + gi * 3 + 1] = z1 - m - l;
  out[NGRP * 3 + gi * 3 + 2] = z2 - m - l;
}

extern "C" void kernel_launch(void* const* d_in, const int* in_sizes, int n_in,
                              void* d_out, int out_size, void* d_ws, size_t ws_size,
                              hipStream_t stream) {
  (void)in_sizes; (void)n_in; (void)out_size; (void)ws_size;
  const float* x     = (const float*)d_in[0];
  const int*   eidx  = (const int*)d_in[1];          // [2,E]
  const int*   batch = (const int*)d_in[2];
  const int* esrc = eidx;
  const int* edst = eidx + NEDGES;

  const float* lin1_W = (const float*)d_in[27];
  const float* lin1_b = (const float*)d_in[28];
  const float* bn5_g  = (const float*)d_in[29];
  const float* bn5_b  = (const float*)d_in[30];
  const float* lin2_W = (const float*)d_in[31];
  const float* lin2_b = (const float*)d_in[32];

  // ---------- workspace layout ----------
  float* ws = (float*)d_ws;
  size_t o = 0;
  float* XL   = ws + o; o += (size_t)NNODES * 128;
  float* XR   = ws + o; o += (size_t)NNODES * 128;
  float* H    = ws + o; o += (size_t)NNODES * 128;       // reused per layer with stride dout
  float* EX   = ws + o; o += (size_t)(NEDGES + NNODES);
  unsigned* MAXV = (unsigned*)(ws + o); o += NNODES;
  float* DEN  = ws + o; o += NNODES;
  float* P    = ws + o; o += (size_t)NGRP * 256;
  float* L1   = ws + o; o += (size_t)NGRP * 128;
  float* MEAN = ws + o; o += 128;
  float* VAR  = ws + o; o += 128;
  float* Z    = ws + o; o += (size_t)NGRP * 3 + 16;      // +pad for alignment
  __bf16* Abf = (__bf16*)(ws + o); o += (size_t)NNODES * 128 / 2;   // bf16 activations
  __bf16* PBL = (__bf16*)(ws + o); o += 32768 / 2;        // packed left weights (<=256x128)
  __bf16* PBR = (__bf16*)(ws + o); o += 16384 / 2;        // packed right weights (<=128x128)

  const int EA = NEDGES + NNODES;
  const unsigned NEG_INF_ENC = 0x007FFFFFu;  // fenc(-inf)

  // zero pooled features
  fill_f32<<<(NGRP * 256 + 255) / 256, 256, 0, stream>>>(P, 0.f, NGRP * 256);

  const float* Xin = x;
  int din = 128;
  const int douts[3]  = {128, 64, 32};
  const int wbase[3]  = {3, 11, 19};
  const int coffA[3]  = {0, 128, 192};
  const int coffB[3]  = {-1, -1, 224};     // p3 concatenated twice

  for (int l = 0; l < 3; ++l) {
    const int dout = douts[l];
    const float* Wl  = (const float*)d_in[wbase[l] + 0];
    const float* bl  = (const float*)d_in[wbase[l] + 1];
    const float* Wr  = (const float*)d_in[wbase[l] + 2];
    const float* br  = (const float*)d_in[wbase[l] + 3];
    const float* att = (const float*)d_in[wbase[l] + 4];
    const float* gb  = (const float*)d_in[wbase[l] + 5];
    const float* bng = (const float*)d_in[wbase[l] + 6];
    const float* bnb = (const float*)d_in[wbase[l] + 7];

    // stage: activations -> bf16, weights -> packed bf16 fragments
    cvt_bf16<<<(NNODES * din + 255) / 256, 256, 0, stream>>>(Xin, Abf, NNODES * din);
    pack_W<<<(din * dout + 255) / 256, 256, 0, stream>>>(Wl, PBL, din, dout);
    pack_W<<<(din * dout + 255) / 256, 256, 0, stream>>>(Wr, PBR, din, dout);

    // XL = X@Wl+bl, XR = X@Wr+br  (WMMA bf16)
    switch (l) {
      case 0: wmma_gemm_t<128,128,true><<<dim3(NNODES/16, 8), 32, 0, stream>>>(Abf, PBL, bl, PBR, br, XL, XR); break;
      case 1: wmma_gemm_t<128, 64,true><<<dim3(NNODES/16, 4), 32, 0, stream>>>(Abf, PBL, bl, PBR, br, XL, XR); break;
      case 2: wmma_gemm_t< 64, 32,true><<<dim3(NNODES/16, 2), 32, 0, stream>>>(Abf, PBL, bl, PBR, br, XL, XR); break;
    }

    fill_u32<<<(NNODES + 255) / 256, 256, 0, stream>>>(MAXV, NEG_INF_ENC, NNODES);
    fill_f32<<<(NNODES + 255) / 256, 256, 0, stream>>>(DEN, 0.f, NNODES);

    edge_logit_max<<<(EA + 7) / 8, 256, 0, stream>>>(XL, XR, att, esrc, edst,
                                                     NEDGES, NNODES, dout, EX, MAXV);
    edge_exp_den<<<(EA + 255) / 256, 256, 0, stream>>>(edst, NEDGES, NNODES, MAXV, EX, DEN);

    const int tot = NNODES * dout;
    init_bias<<<(tot + 255) / 256, 256, 0, stream>>>(H, gb, dout, tot);
    edge_scatter<<<(EA * dout + 255) / 256, 256, 0, stream>>>(XL, EX, DEN, esrc, edst,
                                                              NEDGES, NNODES, dout, H);
    relu_inplace<<<(tot + 255) / 256, 256, 0, stream>>>(H, tot);
    channel_stats<<<dout, 256, 0, stream>>>(H, NNODES, dout, MEAN, VAR);
    bn_norm_pool<<<(tot + 255) / 256, 256, 0, stream>>>(H, NNODES, dout, MEAN, VAR,
                                                        bng, bnb, batch, P, 256,
                                                        coffA[l], coffB[l]);
    Xin = H;      // next layer consumes normalized H
    din = dout;
  }

  // lin1: [256,256] @ [256,128] + b  (WMMA, single matrix)
  cvt_bf16<<<(NGRP * 256 + 255) / 256, 256, 0, stream>>>(P, Abf, NGRP * 256);
  pack_W<<<(256 * 128 + 255) / 256, 256, 0, stream>>>(lin1_W, PBL, 256, 128);
  wmma_gemm_t<256,128,false><<<dim3(NGRP / 16, 8), 32, 0, stream>>>(Abf, PBL, lin1_b,
                                                                    nullptr, nullptr, L1, nullptr);
  relu_inplace<<<(NGRP * 128 + 255) / 256, 256, 0, stream>>>(L1, NGRP * 128);
  channel_stats<<<128, 256, 0, stream>>>(L1, NGRP, 128, MEAN, VAR);
  bn_norm_pool<<<(NGRP * 128 + 255) / 256, 256, 0, stream>>>(L1, NGRP, 128, MEAN, VAR,
                                                             bn5_g, bn5_b, nullptr, nullptr,
                                                             0, 0, -1);
  lin2_kernel<<<(NGRP * 3 + 255) / 256, 256, 0, stream>>>(L1, lin2_W, lin2_b, Z);
  final_out<<<(NGRP + 255) / 256, 256, 0, stream>>>(Z, (float*)d_out);
}
